// LinearKG_34162169872495
// MI455X (gfx1250) — compile-verified, matrix operand order checked by reading, not run on hardware
//
#include <hip/hip_runtime.h>
#include <hip/hip_bf16.h>
#include <stdint.h>

#define DIM_I 256
#define DIM_O 256
#define DIM_R 32
#define BLK   256            // 8 waves
#define MT    128            // rows per workgroup
#define KQ    64             // k-values staged per chunk (2 WMMA k-steps)
#define NCHUNK (DIM_R * (DIM_I / KQ))   // 32 * 4 = 128 chunks

typedef __attribute__((ext_vector_type(16))) __bf16 v16bf;
typedef __attribute__((ext_vector_type(8)))  __bf16 v8bf;
typedef __attribute__((ext_vector_type(8)))  float  v8f;
typedef __attribute__((ext_vector_type(4)))  float  v4f;
typedef __attribute__((ext_vector_type(4)))  int    v4i;

union Frag16 { v16bf v; uint32_t u[8]; };

static __device__ __forceinline__ v16bf join8(v8bf lo, v8bf hi) {
  return __builtin_shufflevector(lo, hi, 0,1,2,3,4,5,6,7,8,9,10,11,12,13,14,15);
}

// ---- CDNA5 async global->LDS copy (16B per lane) with fallback ----
static __device__ __forceinline__ void async_cp16(const __bf16* g, __bf16* l) {
#if __has_builtin(__builtin_amdgcn_global_load_async_to_lds_b128)
  __builtin_amdgcn_global_load_async_to_lds_b128(
      (__attribute__((address_space(1))) v4i*)g,
      (__attribute__((address_space(3))) v4i*)l, 0, 0);
#else
  *(v8bf*)l = *(const v8bf*)g;
#endif
}

static __device__ __forceinline__ void wait_async0() {
#if __has_builtin(__builtin_amdgcn_s_wait_asynccnt)
  __builtin_amdgcn_s_wait_asynccnt(0);
#elif __has_builtin(__builtin_amdgcn_global_load_async_to_lds_b128)
  asm volatile("s_wait_asynccnt 0" ::: "memory");
#endif
}

// ---------- prep: W f32 -> bf16 (row-major [R,O,I]); b -> bT bf16 [O,R] ----------
__global__ void prep_bf16(const float* __restrict__ W, const float* __restrict__ b,
                          __bf16* __restrict__ Wbf, __bf16* __restrict__ bT) {
  int tid = blockIdx.x * blockDim.x + threadIdx.x;
  int nth = gridDim.x * blockDim.x;
  const int NW4 = (DIM_R * DIM_O * DIM_I) / 4;
  for (int i = tid; i < NW4; i += nth) {
    v4f w = ((const v4f*)W)[i];
    union { __bf16 h[4]; uint64_t u; } p;
    p.h[0] = (__bf16)w[0]; p.h[1] = (__bf16)w[1];
    p.h[2] = (__bf16)w[2]; p.h[3] = (__bf16)w[3];
    *(uint64_t*)(Wbf + i * 4) = p.u;
  }
  for (int i = tid; i < DIM_O * DIM_R; i += nth) {
    int o = i / DIM_R, r = i % DIM_R;
    bT[o * DIM_R + r] = (__bf16)b[r * DIM_O + o];
  }
}

// ---------- main fused kernel ----------
__launch_bounds__(BLK)
__global__ void kg_main(const float* __restrict__ x, const float* __restrict__ e,
                        const __bf16* __restrict__ Wbf, const __bf16* __restrict__ bT,
                        float* __restrict__ y, int N) {
  __shared__ __bf16 xs[MT * DIM_I];       // 64 KB: x tile, bf16
  __shared__ __bf16 bs[2][DIM_O * KQ];    // 2 x 32 KB: W_r 64-wide k slice, [o][kk]
  __shared__ float  es[MT * DIM_R];       // 16 KB: edge_attr tile

  const int tid   = threadIdx.x;
  const int lane  = tid & 31;
  const int wave  = tid >> 5;             // 0..7 -> m-tile
  const int row16 = lane & 15;
  const int hi    = lane >> 4;            // 0/1: K-half
  const int wgRow = blockIdx.x * MT;

  // ---- stage x tile (f32 -> bf16, packed b64 stores) ----
  for (int i = tid * 4; i < MT * DIM_I; i += BLK * 4) {
    int lr = i >> 8, col = i & 255;
    int gr = wgRow + lr; if (gr >= N) gr = N - 1;
    v4f v = *(const v4f*)(x + (size_t)gr * DIM_I + col);
    union { __bf16 h[4]; uint64_t u; } p;
    p.h[0] = (__bf16)v[0]; p.h[1] = (__bf16)v[1];
    p.h[2] = (__bf16)v[2]; p.h[3] = (__bf16)v[3];
    *(uint64_t*)&xs[i] = p.u;
  }
  // ---- stage edge_attr tile (f32) ----
  for (int i = tid * 4; i < MT * DIM_R; i += BLK * 4) {
    int lr = i >> 5, col = i & 31;
    int gr = wgRow + lr; if (gr >= N) gr = N - 1;
    *(v4f*)&es[i] = *(const v4f*)(e + (size_t)gr * DIM_R + col);
  }
  __syncthreads();

  auto prefetch_chunk = [&](int c, int buf) {
    int r = c >> 2, kq = c & 3;
    const __bf16* src = Wbf + ((r * DIM_O + tid) * DIM_I + kq * KQ);
    __bf16* dst = &bs[buf][tid * KQ];
    #pragma unroll
    for (int p = 0; p < 8; ++p) async_cp16(src + p * 8, dst + p * 8);
  };
  prefetch_chunk(0, 0);   // overlap first W chunk with bias init

  const int myrow = wave * 16 + row16;

  // ---- bias init via WMMA: acc = E[16x32] x bT-slice[32x16] ----
  Frag16 ef;
  {
    const float* ep = &es[myrow * DIM_R];
    const int kb = hi * 8;
    #pragma unroll
    for (int j = 0; j < 8; ++j) {
      ef.v[j]     = (__bf16)ep[kb + j];
      ef.v[8 + j] = (__bf16)ep[kb + 16 + j];
    }
  }
  v8f acc[16];
  const v8f zro = {0.f, 0.f, 0.f, 0.f, 0.f, 0.f, 0.f, 0.f};
  #pragma unroll
  for (int nt = 0; nt < 16; ++nt) {
    const __bf16* bp = bT + (nt * 16 + row16) * DIM_R + hi * 8;
    v16bf bf = join8(*(const v8bf*)bp, *(const v8bf*)(bp + 16));
    acc[nt] = __builtin_amdgcn_wmma_f32_16x16x32_bf16(
        false, ef.v, false, bf, (short)0, zro, false, false);
  }

  // ---- main loop: 128 chunks of (r, 64 k-values), double-buffered staging ----
  for (int c = 0; c < NCHUNK; ++c) {
    wait_async0();
    __syncthreads();                       // chunk c resident in bs[c&1]
    if (c + 1 < NCHUNK) prefetch_chunk(c + 1, (c + 1) & 1);

    const int r  = c >> 2;
    const int kq = c & 3;

    // relation weight e[myrow, r], packed for v_pk_mul_bf16 (row == lane%16)
    uint32_t ebits;
    {
      union { __bf16 h[2]; uint32_t u; } pk;
      __bf16 eb = (__bf16)es[myrow * DIM_R + r];
      pk.h[0] = eb; pk.h[1] = eb; ebits = pk.u;
    }

    #pragma unroll
    for (int s = 0; s < 2; ++s) {          // two WMMA k-steps per staged chunk
      const __bf16* ap = &xs[myrow * DIM_I + kq * KQ + s * 32 + hi * 8];
      Frag16 a; a.v = join8(*(const v8bf*)ap, *(const v8bf*)(ap + 16));
      Frag16 as;
      #pragma unroll
      for (int j = 0; j < 8; ++j)
        asm("v_pk_mul_bf16 %0, %1, %2" : "=v"(as.u[j]) : "v"(a.u[j]), "v"(ebits));

      #pragma unroll
      for (int nt = 0; nt < 16; ++nt) {
        const __bf16* bp = &bs[c & 1][(nt * 16 + row16) * KQ + s * 32 + hi * 8];
        v16bf bf = join8(*(const v8bf*)bp, *(const v8bf*)(bp + 16));
        acc[nt] = __builtin_amdgcn_wmma_f32_16x16x32_bf16(
            false, as.v, false, bf, (short)0, acc[nt], false, false);
      }
    }
    __syncthreads();                       // protect bs[c&1] before overwrite at c+2
  }

  // ---- epilogue: C layout -> y[N, 256] f32 ----
  #pragma unroll
  for (int nt = 0; nt < 16; ++nt) {
    #pragma unroll
    for (int v = 0; v < 8; ++v) {
      int gr = wgRow + wave * 16 + v + 8 * hi;
      if (gr < N) y[(size_t)gr * DIM_O + nt * 16 + row16] = acc[nt][v];
    }
  }
}

extern "C" void kernel_launch(void* const* d_in, const int* in_sizes, int n_in,
                              void* d_out, int out_size, void* d_ws, size_t ws_size,
                              hipStream_t stream) {
  const float* x = (const float*)d_in[0];   // [N, 256]
  const float* e = (const float*)d_in[1];   // [N, 32]
  const float* W = (const float*)d_in[2];   // [32, 256, 256]
  const float* b = (const float*)d_in[3];   // [32, 256]
  float* y = (float*)d_out;
  const int N = in_sizes[0] / DIM_I;

  __bf16* Wbf = (__bf16*)d_ws;                                   // 4 MB
  __bf16* bT  = (__bf16*)((char*)d_ws + (size_t)DIM_R * DIM_O * DIM_I * 2);

  prep_bf16<<<512, 256, 0, stream>>>(W, b, Wbf, bT);
  const int ng = (N + MT - 1) / MT;
  kg_main<<<ng, BLK, 0, stream>>>(x, e, Wbf, bT, y, N);
}
